// minLSTM_9835475108285
// MI455X (gfx1250) — matrix operand order repaired and checked by reading, compiled, not run on hardware
//
#include <hip/hip_runtime.h>
#include <math.h>

// Problem constants (fixed by the reference)
#define BB 8
#define TT 4096
#define DD 1024
#define HH 1024
#define MM (BB * TT)   // 32768 rows

#define KSTEP 64                   // K elements per pipeline stage
#define LROWU16 72                 // LDS row pitch in u16 (128B data + 16B TDM pad)
#define GATE_TILE_U16 (64 * LROWU16)  // 4608 u16 = 9216 B per gate per stage

typedef __attribute__((ext_vector_type(16))) __bf16   v16bf;
typedef __attribute__((ext_vector_type(8)))  float    v8f;
typedef __attribute__((ext_vector_type(4)))  unsigned u32x4;
typedef __attribute__((ext_vector_type(4)))  unsigned tdm_g0_t;
typedef __attribute__((ext_vector_type(8)))  int      tdm_i8_t;
typedef __attribute__((ext_vector_type(4)))  int      tdm_i4_t;

#if defined(__has_builtin)
#  if __has_builtin(__builtin_amdgcn_tensor_load_to_lds)
#    define USE_TDM 1
#  endif
#endif
#ifndef USE_TDM
#  define USE_TDM 0
#endif

union FragCvt { struct { u32x4 lo; u32x4 hi; } u; v16bf v; };

// A fragment (16x32 bf16), ISA 7.12.2: lane = {m = lane&15, khalf = lane>>4}
// => two 16-byte chunks at k0 + 8*khalf and k0 + 8*khalf + 16.
__device__ __forceinline__ v16bf load_frag_a(const unsigned short* p) {
    FragCvt c;
    c.u.lo = *(const u32x4*)(p);
    c.u.hi = *(const u32x4*)(p + 16);
    return c.v;
}

// B fragment (32x16 bf16): lane = column, two 16-byte chunks at k0 + 8*khalf and k0 + 8*khalf + 16.
__device__ __forceinline__ v16bf load_frag_b(const unsigned short* p) {
    FragCvt c;
    c.u.lo = *(const u32x4*)(p);
    c.u.hi = *(const u32x4*)(p + 16);
    return c.v;
}

__device__ __forceinline__ unsigned short f2bf(float f) {
    unsigned u = __float_as_uint(f);
    u += 0x7FFFu + ((u >> 16) & 1u);   // round-to-nearest-even
    return (unsigned short)(u >> 16);
}

__global__ __launch_bounds__(256) void cvt_f32_bf16(const float* __restrict__ src,
                                                    unsigned short* __restrict__ dst,
                                                    int n4) {
    int i = blockIdx.x * blockDim.x + threadIdx.x;
    if (i >= n4) return;
    const float4 v = *(const float4*)(src + (size_t)i * 4);
    uint2 o;
    o.x = (unsigned)f2bf(v.x) | ((unsigned)f2bf(v.y) << 16);
    o.y = (unsigned)f2bf(v.z) | ((unsigned)f2bf(v.w) << 16);
    *(uint2*)(dst + (size_t)i * 4) = o;
}

#if USE_TDM
// Issue one TDM 2D tile load: 64 rows x 64 u16 from row-major [*, DD] matrix
// into LDS at lds_off, padded 4 dwords per 32 dwords (row pitch 144B).
// Descriptor packing per CDNA5 ISA.
// amdgpu-toolchain (clang-23) 6-arg builtin form.
__device__ __forceinline__ void tdm_load_b_tile(const unsigned short* gptr, unsigned lds_off) {
    const unsigned long long ga = (unsigned long long)(uintptr_t)gptr;
    tdm_g0_t g0;
    g0[0] = 1u;                                         // count=1, user mode
    g0[1] = lds_off;                                    // lds_addr (bytes)
    g0[2] = (unsigned)(ga & 0xFFFFFFFFull);             // global_addr[31:0]
    g0[3] = (unsigned)((ga >> 32) & 0x01FFFFFFull)      // global_addr[56:32]
          | 0x80000000u;                                // type=2 ("image")
    tdm_i8_t g1;
    g1[0] = (int)((1u << 16)      // data_size = 2 bytes
                | (1u << 20)      // pad_enable
                | (4u << 22)      // pad_interval: 32 dwords (128B)
                | (3u << 25));    // pad_amount: 4 dwords (16B)
    g1[1] = (int)((1024u & 0xFFFFu) << 16);   // tensor_dim0[15:0] @ bits 63:48
    g1[2] = (int)((1024u & 0xFFFFu) << 16);   // tensor_dim0 hi=0; tensor_dim1[15:0] @ 95:80
    g1[3] = (int)(64u << 16);                 // tensor_dim1 hi=0; tile_dim0=64 @ 127:112
    g1[4] = (int)(64u);                       // tile_dim1=64; tile_dim2=0
    g1[5] = (int)(1024u);                     // tensor_dim0_stride[31:0]
    g1[6] = 0;                                // stride hi, tensor_dim1_stride lo
    g1[7] = 0;
    const tdm_i4_t gz4 = {0, 0, 0, 0};
    const tdm_i8_t gz8 = {0, 0, 0, 0, 0, 0, 0, 0};
    __builtin_amdgcn_tensor_load_to_lds(g0, g1, gz4, gz4, gz8, 0);
}
#endif

// 24 WMMAs for one 64-K stage: 2 sub-K x 4 n-subtiles x 3 gates.
__device__ __forceinline__ void mma_stage(v8f* acc, const unsigned short* aBase, int k0,
                                          const unsigned short* bs, int l15, int khalf) {
#pragma unroll
    for (int ks = 0; ks < 2; ++ks) {
        const v16bf a = load_frag_a(aBase + k0 + ks * 32);
#pragma unroll
        for (int nt = 0; nt < 4; ++nt) {
            const int row = nt * 16 + l15;
#pragma unroll
            for (int g = 0; g < 3; ++g) {
                const v16bf b = load_frag_b(bs + (size_t)g * GATE_TILE_U16
                                               + row * LROWU16 + ks * 32 + khalf * 8);
                acc[g * 4 + nt] = __builtin_amdgcn_wmma_f32_16x16x32_bf16(
                    false, a, false, b, (short)0, acc[g * 4 + nt], false, false);
            }
        }
    }
}

// Fused 3-gate GEMM + gate epilogue. Block: 8 waves; 128x64 tile x 3 gates.
// B tiles staged in LDS by the Tensor Data Mover, double-buffered.
__global__ __launch_bounds__(256) void gemm_gates(const unsigned short* __restrict__ xb,
                                                  const unsigned short* __restrict__ wb,
                                                  const float* __restrict__ b_f,
                                                  const float* __restrict__ b_i,
                                                  const float* __restrict__ b_h,
                                                  float* __restrict__ fbuf,
                                                  float* __restrict__ vbuf) {
    __shared__ unsigned short Bs[2][3][GATE_TILE_U16];   // 54 KiB

    const int lane  = threadIdx.x & 31;
    const int wave  = threadIdx.x >> 5;
    const int m0    = blockIdx.x * 128 + wave * 16;
    const int n0    = blockIdx.y * 64;
    const int khalf = lane >> 4;
    const int l15   = lane & 15;

    v8f acc[12];
    const v8f vzero = {0.f, 0.f, 0.f, 0.f, 0.f, 0.f, 0.f, 0.f};
#pragma unroll
    for (int i = 0; i < 12; ++i) acc[i] = vzero;

    const unsigned short* aBase = xb + (size_t)(m0 + l15) * DD + khalf * 8;

#if USE_TDM
    if (wave == 0) {
#pragma unroll
        for (int g = 0; g < 3; ++g)
            tdm_load_b_tile(wb + (size_t)(g * HH + n0) * DD,
                            (unsigned)(uintptr_t)&Bs[0][g][0]);
    }
    for (int k0 = 0; k0 < DD; k0 += KSTEP) {
        const int cur = (k0 >> 6) & 1;
        if (wave == 0) __builtin_amdgcn_s_wait_tensorcnt(0);
        __syncthreads();
        if (wave == 0 && (k0 + KSTEP) < DD) {
#pragma unroll
            for (int g = 0; g < 3; ++g)
                tdm_load_b_tile(wb + (size_t)(g * HH + n0) * DD + (k0 + KSTEP),
                                (unsigned)(uintptr_t)&Bs[cur ^ 1][g][0]);
        }
        mma_stage(acc, aBase, k0, &Bs[cur][0][0], l15, khalf);
        __syncthreads();
    }
#else
    // Fallback: cooperative global->LDS copy (single-buffered).
    for (int k0 = 0; k0 < DD; k0 += KSTEP) {
        for (int it = threadIdx.x; it < 3 * 64 * 8; it += 256) {
            const int g   = it >> 9;           // / 512
            const int rem = it & 511;
            const int r   = rem >> 3;
            const int c8  = rem & 7;
            const u32x4 d = *(const u32x4*)(wb + (size_t)(g * HH + n0 + r) * DD + k0 + c8 * 8);
            *(u32x4*)&Bs[0][g][r * LROWU16 + c8 * 8] = d;
        }
        __syncthreads();
        mma_stage(acc, aBase, k0, &Bs[0][0][0], l15, khalf);
        __syncthreads();
    }
#endif

    // Epilogue: C/D layout — VGPR r, lanes 0-15: (m0+r, n), lanes 16-31: (m0+r+8, n)
#pragma unroll
    for (int nt = 0; nt < 4; ++nt) {
        const int n  = n0 + nt * 16 + l15;
        const float bfv = b_f[n];
        const float biv = b_i[n];
        const float bhv = b_h[n];
#pragma unroll
        for (int r = 0; r < 8; ++r) {
            const int m = m0 + r + khalf * 8;
            const float zf = acc[0 * 4 + nt][r] + bfv;
            const float zi = acc[1 * 4 + nt][r] + biv;
            const float zh = acc[2 * 4 + nt][r] + bhv;
            // softplus(-z) = max(-z,0) + log1p(exp(-|z|))
            const float spf  = fmaxf(-zf, 0.f) + log1pf(__expf(-fabsf(zf)));
            const float spi  = fmaxf(-zi, 0.f) + log1pf(__expf(-fabsf(zi)));
            const float diff = spf - spi;
            const float fg   = 1.f / (1.f + __expf(diff));   // sigmoid(-diff)
            const float ig   = 1.f - fg;                     // sigmoid(diff)
            const float gv   = (zh >= 0.f) ? (zh + 0.5f) : 1.f / (1.f + __expf(-zh));
            const size_t oi  = (size_t)m * HH + n;
            fbuf[oi] = fg;
            vbuf[oi] = ig * gv;
        }
    }
}

// Linear-space scan: h_t = f_t * h_{t-1} + v_t, h_init = g(h0).
__global__ __launch_bounds__(256) void scan_kernel(const float* __restrict__ fbuf,
                                                   const float* __restrict__ vbuf,
                                                   const float* __restrict__ h0,
                                                   float* __restrict__ out) {
    const int tid = blockIdx.x * blockDim.x + threadIdx.x;
    if (tid >= BB * HH) return;
    const int b = tid >> 10;        // / HH
    const int h = tid & (HH - 1);   // % HH
    const float x0 = h0[b * HH + h];
    float hc = (x0 >= 0.f) ? (x0 + 0.5f) : 1.f / (1.f + __expf(-x0));
    size_t idx = (size_t)b * TT * HH + h;
#pragma unroll 4
    for (int t = 0; t < TT; ++t, idx += HH) {
        hc = fmaf(fbuf[idx], hc, vbuf[idx]);
        out[idx] = hc;
    }
}

extern "C" void kernel_launch(void* const* d_in, const int* in_sizes, int n_in,
                              void* d_out, int out_size, void* d_ws, size_t ws_size,
                              hipStream_t stream) {
    const float* x   = (const float*)d_in[0];
    const float* h0  = (const float*)d_in[1];
    const float* W_f = (const float*)d_in[2];
    const float* b_f = (const float*)d_in[3];
    const float* W_i = (const float*)d_in[4];
    const float* b_i = (const float*)d_in[5];
    const float* W_h = (const float*)d_in[6];
    const float* b_h = (const float*)d_in[7];
    float* out = (float*)d_out;

    // Workspace layout
    char* ws = (char*)d_ws;
    const size_t xb_bytes = (size_t)MM * DD * 2;        // 64 MiB
    const size_t wb_bytes = (size_t)3 * HH * DD * 2;    // 6 MiB
    const size_t fv_bytes = (size_t)MM * HH * 4;        // 128 MiB each
    unsigned short* xb = (unsigned short*)ws;
    unsigned short* wb = (unsigned short*)(ws + xb_bytes);
    float* fbuf = (float*)(ws + xb_bytes + wb_bytes);
    float* vbuf = (float*)(ws + xb_bytes + wb_bytes + fv_bytes);

    // 1) fp32 -> bf16 conversions (x and packed [3H, D] weights)
    {
        const int n4x = (MM * DD) / 4;
        cvt_f32_bf16<<<n4x / 256, 256, 0, stream>>>(x, xb, n4x);
        const int n4w = (HH * DD) / 4;
        cvt_f32_bf16<<<n4w / 256, 256, 0, stream>>>(W_f, wb,                     n4w);
        cvt_f32_bf16<<<n4w / 256, 256, 0, stream>>>(W_i, wb + (size_t)HH * DD,   n4w);
        cvt_f32_bf16<<<n4w / 256, 256, 0, stream>>>(W_h, wb + (size_t)2*HH*DD,   n4w);
    }

    // 2) Fused WMMA GEMM (3 gates, TDM-staged B tiles) -> f, v
    dim3 grid(MM / 128, HH / 64);    // 256 x 16
    gemm_gates<<<grid, 256, 0, stream>>>(xb, wb, b_f, b_i, b_h, fbuf, vbuf);

    // 3) Recurrence scan -> out
    scan_kernel<<<(BB * HH) / 256, 256, 0, stream>>>(fbuf, vbuf, h0, out);
}
